// GraphConvClusterPool_15247133901688
// MI455X (gfx1250) — compile-verified, matrix operand order checked by reading, not run on hardware
//
#include <hip/hip_runtime.h>
#include <hip/hip_bf16.h>
#include <math.h>

#define N_NODES_C 50000
#define N_EDGES_C 400000
#define P_NODES_C (N_NODES_C + 1)
#define CC_ITERS_C 25
#define HBITS 20
#define HSIZE_C (1u << HBITS)

typedef __bf16 bf16_t;
typedef bf16_t v16bf __attribute__((ext_vector_type(16)));
typedef float v8f __attribute__((ext_vector_type(8)));

__device__ __forceinline__ unsigned short bf_bits(float x) {
  unsigned u = __float_as_uint(x);
  unsigned r = u + 0x7FFFu + ((u >> 16) & 1u);
  return (unsigned short)(r >> 16);
}

// Pre-convert W (K x M, f32, row-major) into Wt (M x K, bf16): Wt[m][k] = W[k][m].
__global__ void k_w_to_bf16t(const float* __restrict__ W, unsigned short* __restrict__ Wt,
                             int K, int M) {
  int idx = blockIdx.x * blockDim.x + threadIdx.x;
  if (idx >= K * M) return;
  int m = idx / K;
  int k = idx - m * K;
  Wt[idx] = bf_bits(W[(size_t)k * M + m]);
}

// One 16x16x32 bf16 WMMA step at K-offset kt.
__device__ __forceinline__ v8f wmma_step(const float* __restrict__ Arow,
                                         const unsigned short* __restrict__ Wtrow,
                                         int kt, int half8, v8f acc) {
  union { v16bf v; unsigned short u[16]; uint4 q[2]; } afrag, bfrag;
  // A: two contiguous 8-float runs per lane (K = kt+half8.. and kt+16+half8..)
  float av[16];
  *(float4*)&av[0]  = *(const float4*)(Arow + kt + half8);
  *(float4*)&av[4]  = *(const float4*)(Arow + kt + half8 + 4);
  *(float4*)&av[8]  = *(const float4*)(Arow + kt + 16 + half8);
  *(float4*)&av[12] = *(const float4*)(Arow + kt + 16 + half8 + 4);
#pragma unroll
  for (int t = 0; t < 16; ++t) afrag.u[t] = bf_bits(av[t]);
  // B: Wt row for this column is contiguous in K -> two b128 loads of 8 bf16 each
  bfrag.q[0] = *(const uint4*)(Wtrow + kt + half8);
  bfrag.q[1] = *(const uint4*)(Wtrow + kt + 16 + half8);
  return __builtin_amdgcn_wmma_f32_16x16x32_bf16(false, afrag.v, false, bfrag.v,
                                                 (short)0, acc, false, false);
}

// ---------------------------------------------------------------------------
// WMMA GEMM: C[n x M] = act(A[n x K] @ W[K x M] + bias). M multiple of 16,
// K multiple of 64. One wave per 16x16 output tile, 4 waves per block.
// A is f32 (converted on the fly, RNE); W pre-converted to bf16 transposed.
// Out-of-range rows are handled by clamping (their results are never stored).
// ---------------------------------------------------------------------------
__global__ void gemm_wmma_bf16(const float* __restrict__ A, int lda,
                               const unsigned short* __restrict__ Wt,
                               const float* __restrict__ bias,
                               float* __restrict__ C, int ldc, int colOff,
                               int n, int K, int M, int relu) {
  const int lane = threadIdx.x & 31;
  const int wave = threadIdx.x >> 5;
  const int colTiles = M >> 4;
  const int rowTiles = (n + 15) >> 4;
  const int tile = blockIdx.x * 4 + wave;
  if (tile >= rowTiles * colTiles) return;
  const int rowTile = tile / colTiles;
  const int colTile = tile - rowTile * colTiles;
  const int m = lane & 15;
  const int half8 = (lane >> 4) * 8;
  const int row = min(rowTile * 16 + m, n - 1);  // clamp: extra rows never stored
  const int col = colTile * 16 + m;
  const float* Arow = A + (size_t)row * lda;
  const unsigned short* Wtrow = Wt + (size_t)col * K;

  v8f acc = {0.f, 0.f, 0.f, 0.f, 0.f, 0.f, 0.f, 0.f};
  for (int kt = 0; kt < K; kt += 64) {
    if (kt + 64 < K) {
      __builtin_prefetch(Arow + kt + 64, 0, 1);    // global_prefetch_b8: next A slab
    }
    acc = wmma_step(Arow, Wtrow, kt, half8, acc);
    acc = wmma_step(Arow, Wtrow, kt + 32, half8, acc);
  }

  // epilogue: bias hoisted once per wave; full-tile fast path (tail only in the
  // last row tile).
  const float bv = bias ? bias[col] : 0.f;
  const int rbase = rowTile * 16 + (half8 & 8);    // C/D layout: M = vgpr + 8*half
  float* Cc = C + colOff + col;
  if (rowTile * 16 + 16 <= n) {
#pragma unroll
    for (int v = 0; v < 8; ++v) {
      float val = acc[v] + bv;
      if (relu) val = fmaxf(val, 0.f);
      Cc[(size_t)(rbase + v) * ldc] = val;
    }
  } else {
#pragma unroll
    for (int v = 0; v < 8; ++v) {
      if (rbase + v < n) {
        float val = acc[v] + bv;
        if (relu) val = fmaxf(val, 0.f);
        Cc[(size_t)(rbase + v) * ldc] = val;
      }
    }
  }
}

// --------------------------- utility kernels -------------------------------
__global__ void k_split_edges(const int* __restrict__ ei, int* __restrict__ src,
                              int* __restrict__ dst, int E) {
  int e = blockIdx.x * blockDim.x + threadIdx.x;
  if (e < E) { src[e] = ei[2 * e]; dst[e] = ei[2 * e + 1]; }
}

__global__ void k_fill_f32(float* __restrict__ p, float v, long long n) {
  long long i = (long long)blockIdx.x * blockDim.x + threadIdx.x;
  if (i < n) p[i] = v;
}

__global__ void k_fill_u64(unsigned long long* __restrict__ p,
                           unsigned long long v, long long n) {
  long long i = (long long)blockIdx.x * blockDim.x + threadIdx.x;
  if (i < n) p[i] = v;
}

__global__ void k_iota(int* __restrict__ p, int n) {
  int i = blockIdx.x * blockDim.x + threadIdx.x;
  if (i < n) p[i] = i;
}

__global__ void k_copy_i32(const int* __restrict__ a, int* __restrict__ b, int n) {
  int i = blockIdx.x * blockDim.x + threadIdx.x;
  if (i < n) b[i] = a[i];
}

__global__ void k_copy_cols(const float* __restrict__ in, int ldi,
                            float* __restrict__ out, int ldo, int colOff,
                            int n, int f) {
  long long idx = (long long)blockIdx.x * blockDim.x + threadIdx.x;
  if (idx >= (long long)n * f) return;
  int i = (int)(idx / f);
  int j = (int)(idx - (long long)i * f);
  out[(size_t)i * ldo + colOff + j] = in[(size_t)i * ldi + j];
}

// ------------------------------ GCN kernels --------------------------------
__global__ void k_deg_scatter(const int* __restrict__ dst, float* __restrict__ deg, int E) {
  int e = blockIdx.x * blockDim.x + threadIdx.x;
  if (e < E) atomicAdd(&deg[dst[e]], 1.0f);
}

__global__ void k_rsqrt(const float* __restrict__ deg, float* __restrict__ dinv, int n) {
  int i = blockIdx.x * blockDim.x + threadIdx.x;
  if (i < n) dinv[i] = rsqrtf(deg[i]);
}

__global__ void k_gcn_scatter(const float* __restrict__ h, const int* __restrict__ src,
                              const int* __restrict__ dst, const float* __restrict__ dinv,
                              float* __restrict__ agg, int E) {
  long long idx = (long long)blockIdx.x * blockDim.x + threadIdx.x;
  int e = (int)(idx >> 5);
  if (e >= E) return;
  int lane = (int)(idx & 31);
  int s = src[e], d = dst[e];
  float coef = dinv[s] * dinv[d];
  const float* hs = h + (size_t)s * 128;
  float* ad = agg + (size_t)d * 128;
#pragma unroll
  for (int j = 0; j < 4; ++j) {
    int fidx = lane + 32 * j;
    atomicAdd(&ad[fidx], hs[fidx] * coef);
  }
}

__global__ void k_gcn_finalize(const float* __restrict__ agg, const float* __restrict__ hW,
                               const float* __restrict__ deg, const float* __restrict__ bias,
                               float* __restrict__ out, int ldo, int colOff, int n) {
  long long idx = (long long)blockIdx.x * blockDim.x + threadIdx.x;
  if (idx >= (long long)n * 128) return;
  int i = (int)(idx >> 7);
  int j = (int)(idx & 127);
  float v = agg[idx] + hW[idx] / deg[i] + bias[j];
  out[(size_t)i * ldo + colOff + j] = fmaxf(v, 0.f);
}

// ------------------------------ pool kernels -------------------------------
__global__ void k_pool_node_scores(const float* __restrict__ x, int f,
                                   const float* __restrict__ w,
                                   float* __restrict__ pa, float* __restrict__ pb, int n) {
  int i = blockIdx.x;
  if (i >= n) return;
  float s1 = 0.f, s2 = 0.f;
  for (int j = threadIdx.x; j < f; j += blockDim.x) {
    float v = x[(size_t)i * f + j];
    s1 += v * w[j];
    s2 += v * w[f + j];
  }
  __shared__ float r1[128];
  __shared__ float r2[128];
  r1[threadIdx.x] = s1; r2[threadIdx.x] = s2;
  __syncthreads();
  for (int s = 64; s > 0; s >>= 1) {
    if (threadIdx.x < s) { r1[threadIdx.x] += r1[threadIdx.x + s];
                           r2[threadIdx.x] += r2[threadIdx.x + s]; }
    __syncthreads();
  }
  if (threadIdx.x == 0) { pa[i] = r1[0]; pb[i] = r2[0]; }
}

__global__ void k_edge_score(const float* __restrict__ pa, const float* __restrict__ pb,
                             const float* __restrict__ bptr,
                             const int* __restrict__ src, const int* __restrict__ dst,
                             float* __restrict__ score, int* __restrict__ sel, int E) {
  int e = blockIdx.x * blockDim.x + threadIdx.x;
  if (e >= E) return;
  int s = src[e], d = dst[e];
  float sc = tanhf(pa[s] + pb[d] + bptr[0]);
  score[e] = sc;
  sel[e] = (sc > 0.f && s != d) ? 1 : 0;
}

__global__ void k_edge_min(const int* __restrict__ src, const int* __restrict__ dst,
                           const int* __restrict__ sel, const int* __restrict__ lblOld,
                           int* __restrict__ lblNew, int E) {
  int e = blockIdx.x * blockDim.x + threadIdx.x;
  if (e >= E || !sel[e]) return;
  int s = src[e], d = dst[e];
  int m = min(lblOld[s], lblOld[d]);
  atomicMin(&lblNew[s], m);
  atomicMin(&lblNew[d], m);
}

__global__ void k_jump(const int* __restrict__ in, int* __restrict__ out, int n) {
  int i = blockIdx.x * blockDim.x + threadIdx.x;
  if (i < n) out[i] = in[in[i]];
}

__global__ void k_csize(const int* __restrict__ cl, float* __restrict__ csize, int n) {
  int i = blockIdx.x * blockDim.x + threadIdx.x;
  if (i < n) atomicAdd(&csize[cl[i]], 1.0f);
}

__global__ void k_edge_stats(const int* __restrict__ sel, const float* __restrict__ score,
                             const int* __restrict__ cl, const int* __restrict__ src,
                             float* __restrict__ ssum, float* __restrict__ scnt, int E) {
  int e = blockIdx.x * blockDim.x + threadIdx.x;
  if (e >= E || !sel[e]) return;
  int c = cl[src[e]];
  atomicAdd(&ssum[c], score[e]);
  atomicAdd(&scnt[c], 1.0f);
}

__global__ void k_cscore(const float* __restrict__ csize, const float* __restrict__ ssum,
                         const float* __restrict__ scnt, float* __restrict__ cscore, int n) {
  int i = blockIdx.x * blockDim.x + threadIdx.x;
  if (i < n) cscore[i] = (csize[i] > 1.0f) ? (ssum[i] / fmaxf(scnt[i], 1.0f)) : 1.0f;
}

__global__ void k_newx_scatter(const float* __restrict__ x, const int* __restrict__ cl,
                               float* __restrict__ nx, int n, int f) {
  long long idx = (long long)blockIdx.x * blockDim.x + threadIdx.x;
  if (idx >= (long long)n * f) return;
  int i = (int)(idx / f);
  int j = (int)(idx - (long long)i * f);
  atomicAdd(&nx[(size_t)cl[i] * f + j], x[idx]);
}

__global__ void k_scale_rows(float* __restrict__ nx, const float* __restrict__ cscore,
                             int n, int f) {
  long long idx = (long long)blockIdx.x * blockDim.x + threadIdx.x;
  if (idx >= (long long)n * f) return;
  int i = (int)(idx / f);
  nx[idx] *= cscore[i];
}

__global__ void k_remap_edges(const int* __restrict__ src, const int* __restrict__ dst,
                              const int* __restrict__ cl, int* __restrict__ nsrc,
                              int* __restrict__ ndst, unsigned long long* __restrict__ htab,
                              unsigned hmask, int E, int dummy) {
  int e = blockIdx.x * blockDim.x + threadIdx.x;
  if (e >= E) return;
  int a = cl[src[e]];
  int c = cl[dst[e]];
  int os = dummy, od = dummy;
  if (a != c) {
    unsigned long long key = ((unsigned long long)(unsigned)a << 32) | (unsigned)c;
    unsigned h = (unsigned)((key * 0x9E3779B97F4A7C15ull) >> 44) & hmask;
    for (unsigned probe = 0; probe <= hmask; ++probe) {
      unsigned long long prev = atomicCAS(&htab[h], ~0ull, key);
      if (prev == ~0ull) { os = a; od = c; break; }  // first occurrence -> keep
      if (prev == key) break;                        // duplicate -> drop
      h = (h + 1) & hmask;
    }
  }
  nsrc[e] = os; ndst[e] = od;
}

__global__ void k_unpool(const int* __restrict__ cm1, const int* __restrict__ cm2,
                         const int* __restrict__ cm3, const float* __restrict__ g,
                         float* __restrict__ out, int ldo, int colOff, int n) {
  long long idx = (long long)blockIdx.x * blockDim.x + threadIdx.x;
  if (idx >= (long long)n * 128) return;
  int i = (int)(idx >> 7);
  int j = (int)(idx & 127);
  int r = cm3[cm2[cm1[i]]];
  out[(size_t)i * ldo + colOff + j] = g[(size_t)r * 128 + j];
}

__global__ void k_fc2_sigmoid(const float* __restrict__ x, int f,
                              const float* __restrict__ w, const float* __restrict__ bptr,
                              float* __restrict__ out, int n) {
  int i = blockIdx.x;
  if (i >= n) return;
  float s = 0.f;
  for (int j = threadIdx.x; j < f; j += blockDim.x) s += x[(size_t)i * f + j] * w[j];
  __shared__ float r[128];
  r[threadIdx.x] = s;
  __syncthreads();
  for (int k = 64; k > 0; k >>= 1) {
    if (threadIdx.x < k) r[threadIdx.x] += r[threadIdx.x + k];
    __syncthreads();
  }
  if (threadIdx.x == 0) out[i] = 1.f / (1.f + expf(-(r[0] + bptr[0])));
}

// ---------------------------------------------------------------------------
static inline unsigned gdiv(long long a, long long b) { return (unsigned)((a + b - 1) / b); }

extern "C" void kernel_launch(void* const* d_in, const int* in_sizes, int n_in,
                              void* d_out, int out_size, void* d_ws, size_t ws_size,
                              hipStream_t stream) {
  (void)in_sizes; (void)n_in; (void)out_size; (void)ws_size;
  const int N = N_NODES_C, E = N_EDGES_C, P = P_NODES_C;

  const float* x   = (const float*)d_in[0];
  const int*   ei  = (const int*)d_in[1];
  const float* cw1 = (const float*)d_in[2];  const float* cb1 = (const float*)d_in[3];
  const float* pw1 = (const float*)d_in[4];  const float* pb1 = (const float*)d_in[5];
  const float* cw4 = (const float*)d_in[6];  const float* cb4 = (const float*)d_in[7];
  const float* pw2 = (const float*)d_in[8];  const float* pb2 = (const float*)d_in[9];
  const float* cw5 = (const float*)d_in[10]; const float* cb5 = (const float*)d_in[11];
  const float* pw3 = (const float*)d_in[12]; const float* pb3 = (const float*)d_in[13];
  const float* cw6 = (const float*)d_in[14]; const float* cb6 = (const float*)d_in[15];
  const float* cw7 = (const float*)d_in[16]; const float* cb7 = (const float*)d_in[17];
  const float* fw1 = (const float*)d_in[18]; const float* fb1 = (const float*)d_in[19];
  const float* fw2 = (const float*)d_in[20]; const float* fb2 = (const float*)d_in[21];
  float* out = (float*)d_out;

  // ---- bump allocator over workspace ----
  char* base = (char*)d_ws;
  size_t off = 0;
  auto alloc = [&](size_t bytes) -> char* {
    off = (off + 255) & ~(size_t)255;
    char* p = base + off;
    off += bytes;
    return p;
  };

  int* src0 = (int*)alloc((size_t)E * 4); int* dst0 = (int*)alloc((size_t)E * 4);
  int* src1 = (int*)alloc((size_t)E * 4); int* dst1 = (int*)alloc((size_t)E * 4);
  int* src2 = (int*)alloc((size_t)E * 4); int* dst2 = (int*)alloc((size_t)E * 4);
  int* src3 = (int*)alloc((size_t)E * 4); int* dst3 = (int*)alloc((size_t)E * 4);
  float* hW   = (float*)alloc((size_t)P * 128 * 4);
  float* agg  = (float*)alloc((size_t)P * 128 * 4);
  float* deg  = (float*)alloc((size_t)P * 4);
  float* dinv = (float*)alloc((size_t)P * 4);
  float* cat0 = (float*)alloc((size_t)N * 256 * 4);
  float* h1   = (float*)alloc((size_t)P * 256 * 4);
  float* cat1 = (float*)alloc((size_t)P * 384 * 4);
  float* h2   = (float*)alloc((size_t)P * 384 * 4);
  float* cat2 = (float*)alloc((size_t)P * 512 * 4);
  float* h3   = (float*)alloc((size_t)P * 512 * 4);
  float* g6   = (float*)alloc((size_t)P * 128 * 4);
  float* pa   = (float*)alloc((size_t)P * 4);
  float* pb   = (float*)alloc((size_t)P * 4);
  float* score = (float*)alloc((size_t)E * 4);
  int*   sel   = (int*)alloc((size_t)E * 4);
  int* lblB = (int*)alloc((size_t)P * 4);
  int* lblC = (int*)alloc((size_t)P * 4);
  int* cm1  = (int*)alloc((size_t)N * 4);
  int* cm2  = (int*)alloc((size_t)P * 4);
  int* cm3  = (int*)alloc((size_t)P * 4);
  float* csize  = (float*)alloc((size_t)P * 4);
  float* ssum   = (float*)alloc((size_t)P * 4);
  float* scnt   = (float*)alloc((size_t)P * 4);
  float* cscore = (float*)alloc((size_t)P * 4);
  unsigned short* WtBuf = (unsigned short*)alloc((size_t)512 * 128 * 2);  // max K=512
  unsigned long long* htab = (unsigned long long*)alloc((size_t)HSIZE_C * 8);
  // region reuse (producers fully consumed before reuse; stream-ordered)
  float* cat3 = cat0;
  float* cat4 = cat1;
  float* cat5 = cat2;

  auto launch_gemm = [&](const float* A, int K, const float* W, const float* bias,
                         float* C, int ldc, int colOff, int n, int relu) {
    const int M = 128;
    k_w_to_bf16t<<<gdiv((long long)K * M, 256), 256, 0, stream>>>(W, WtBuf, K, M);
    int tiles = ((n + 15) / 16) * (M / 16);
    gemm_wmma_bf16<<<gdiv(tiles, 4), 128, 0, stream>>>(A, K, WtBuf, bias, C, ldc,
                                                       colOff, n, K, M, relu);
  };

  auto run_gcn = [&](const float* A, int K, const float* W, const float* bias,
                     const int* src, const int* dst, int n,
                     float* outBuf, int ldo, int colOff) {
    launch_gemm(A, K, W, nullptr, hW, 128, 0, n, 0);                 // hW = A @ W
    k_fill_f32<<<gdiv(n, 256), 256, 0, stream>>>(deg, 1.0f, n);
    k_deg_scatter<<<gdiv(E, 256), 256, 0, stream>>>(dst, deg, E);
    k_rsqrt<<<gdiv(n, 256), 256, 0, stream>>>(deg, dinv, n);
    k_fill_f32<<<gdiv((long long)n * 128, 256), 256, 0, stream>>>(agg, 0.f, (long long)n * 128);
    k_gcn_scatter<<<gdiv((long long)E * 32, 256), 256, 0, stream>>>(hW, src, dst, dinv, agg, E);
    k_gcn_finalize<<<gdiv((long long)n * 128, 256), 256, 0, stream>>>(agg, hW, deg, bias,
                                                                      outBuf, ldo, colOff, n);
  };

  auto run_pool = [&](const float* X, int f, int n, const float* pw, const float* pbS,
                      const int* src, const int* dst,
                      int* cluster, float* newX, int* nsrc, int* ndst) {
    const int dummy = P - 1;
    k_pool_node_scores<<<n, 128, 0, stream>>>(X, f, pw, pa, pb, n);
    k_edge_score<<<gdiv(E, 256), 256, 0, stream>>>(pa, pb, pbS, src, dst, score, sel, E);
    k_iota<<<gdiv(n, 256), 256, 0, stream>>>(cluster, n);
    for (int it = 0; it < CC_ITERS_C; ++it) {
      k_copy_i32<<<gdiv(n, 256), 256, 0, stream>>>(cluster, lblB, n);
      k_edge_min<<<gdiv(E, 256), 256, 0, stream>>>(src, dst, sel, cluster, lblB, E);
      k_jump<<<gdiv(n, 256), 256, 0, stream>>>(lblB, lblC, n);
      k_jump<<<gdiv(n, 256), 256, 0, stream>>>(lblC, cluster, n);
    }
    k_fill_f32<<<gdiv(P, 256), 256, 0, stream>>>(csize, 0.f, P);
    k_fill_f32<<<gdiv(P, 256), 256, 0, stream>>>(ssum, 0.f, P);
    k_fill_f32<<<gdiv(P, 256), 256, 0, stream>>>(scnt, 0.f, P);
    k_csize<<<gdiv(n, 256), 256, 0, stream>>>(cluster, csize, n);
    k_edge_stats<<<gdiv(E, 256), 256, 0, stream>>>(sel, score, cluster, src, ssum, scnt, E);
    k_cscore<<<gdiv(P, 256), 256, 0, stream>>>(csize, ssum, scnt, cscore, P);
    k_fill_f32<<<gdiv((long long)P * f, 256), 256, 0, stream>>>(newX, 0.f, (long long)P * f);
    k_newx_scatter<<<gdiv((long long)n * f, 256), 256, 0, stream>>>(X, cluster, newX, n, f);
    k_scale_rows<<<gdiv((long long)P * f, 256), 256, 0, stream>>>(newX, cscore, P, f);
    k_fill_u64<<<gdiv(HSIZE_C, 256), 256, 0, stream>>>(htab, ~0ull, (long long)HSIZE_C);
    k_remap_edges<<<gdiv(E, 256), 256, 0, stream>>>(src, dst, cluster, nsrc, ndst,
                                                    htab, HSIZE_C - 1, E, dummy);
  };

  // ---- pipeline ----
  k_split_edges<<<gdiv(E, 256), 256, 0, stream>>>(ei, src0, dst0, E);

  // level 0: gcn1 -> cat0 = [x | relu(gcn)]
  run_gcn(x, 128, cw1, cb1, src0, dst0, N, cat0, 256, 128);
  k_copy_cols<<<gdiv((long long)N * 128, 256), 256, 0, stream>>>(x, 128, cat0, 256, 0, N, 128);

  // pool 1
  run_pool(cat0, 256, N, pw1, pb1, src0, dst0, cm1, h1, src1, dst1);

  // level 1: gcn4 -> cat1 = [h1 | relu(gcn)]
  run_gcn(h1, 256, cw4, cb4, src1, dst1, P, cat1, 384, 256);
  k_copy_cols<<<gdiv((long long)P * 256, 256), 256, 0, stream>>>(h1, 256, cat1, 384, 0, P, 256);

  // pool 2
  run_pool(cat1, 384, P, pw2, pb2, src1, dst1, cm2, h2, src2, dst2);

  // level 2: gcn5 -> cat2 = [h2 | relu(gcn)]
  run_gcn(h2, 384, cw5, cb5, src2, dst2, P, cat2, 512, 384);
  k_copy_cols<<<gdiv((long long)P * 384, 256), 256, 0, stream>>>(h2, 384, cat2, 512, 0, P, 384);

  // pool 3
  run_pool(cat2, 512, P, pw3, pb3, src2, dst2, cm3, h3, src3, dst3);

  // level 3: gcn6 -> g6 (P x 128), relu
  run_gcn(h3, 512, cw6, cb6, src3, dst3, P, g6, 128, 0);

  // unpool: cat3 = [x | g6[cm3[cm2[cm1]]]]
  k_copy_cols<<<gdiv((long long)N * 128, 256), 256, 0, stream>>>(x, 128, cat3, 256, 0, N, 128);
  k_unpool<<<gdiv((long long)N * 128, 256), 256, 0, stream>>>(cm1, cm2, cm3, g6, cat3, 256, 128, N);

  // gcn7 on original graph -> cat4 = [x | relu(gcn)]
  run_gcn(cat3, 256, cw7, cb7, src0, dst0, N, cat4, 256, 128);
  k_copy_cols<<<gdiv((long long)N * 128, 256), 256, 0, stream>>>(x, 128, cat4, 256, 0, N, 128);

  // fc1 (WMMA, fused bias+relu) -> cat5 = [x | relu(cat4 @ fw1 + fb1)]
  launch_gemm(cat4, 256, fw1, fb1, cat5, 256, 128, N, 1);
  k_copy_cols<<<gdiv((long long)N * 128, 256), 256, 0, stream>>>(x, 128, cat5, 256, 0, N, 128);

  // fc2 + sigmoid -> d_out
  k_fc2_sigmoid<<<N, 128, 0, stream>>>(cat5, 256, fw2, fb2, out, N);
}
